// ContextMessageBlock_23802708755005
// MI455X (gfx1250) — matrix operand order repaired and checked by reading, not compiled, not run on hardware
//
#include <hip/hip_runtime.h>
#include <hip/hip_bf16.h>

typedef __attribute__((ext_vector_type(16))) __bf16 v16bf;
typedef __attribute__((ext_vector_type(8)))  float  v8f;

#define HID 128
#define NUM_RBF 32
#define CUTOFF 6.0f
#define DPAD 448          // 417 padded up to multiple of 32
#define APAD 456          // LDS row pitch for edge A tile (16B aligned, conflict-free)
#define HPAD 136          // LDS row pitch for 128-wide bf16 tiles
#define NPAD 264          // LDS row pitch for 256-wide bf16 tile (node kernel)
#define UPAD 132          // LDS row pitch for f32 update tile
#define ME   64           // edges per block (4 M-tiles of 16)
#define MN   64           // nodes per block (4 M-tiles of 16)

static __device__ __forceinline__ __bf16 f2bf(float f) {
    unsigned u = __float_as_uint(f);
    u += 0x7FFFu + ((u >> 16) & 1u);          // round-to-nearest-even
    unsigned short s = (unsigned short)(u >> 16);
    __bf16 r;
    __builtin_memcpy(&r, &s, sizeof(r));
    return r;
}

// bf16(a) in low 16 bits, bf16(b) in high 16 bits
static __device__ __forceinline__ unsigned pack2(float a, float b) {
    unsigned ua = __float_as_uint(a); ua += 0x7FFFu + ((ua >> 16) & 1u);
    unsigned ub = __float_as_uint(b); ub += 0x7FFFu + ((ub >> 16) & 1u);
    return (ua >> 16) | (ub & 0xFFFF0000u);
}

static __device__ __forceinline__ float silu(float x) {
    return x * (1.0f / (1.0f + __expf(-x)));
}

// ---------------------------------------------------------------- zero
__global__ void zero_kernel(float* sums, float* counts, int nSums, int nCounts) {
    int i = blockIdx.x * blockDim.x + threadIdx.x;
    if (i < nSums) sums[i] = 0.0f;
    if (i < nCounts) counts[i] = 0.0f;
}

// ------------------------------------------------------- weight prep (f32 -> bf16, transposed+padded)
__global__ void prep_kernel(const float* mw1, const float* mw2,
                            const float* uw1, const float* uw2,
                            __bf16* mw1t, __bf16* mw2t, __bf16* uw1t, __bf16* uw2t) {
    int idx = blockIdx.x * blockDim.x + threadIdx.x;
    if (idx < 128 * DPAD) {
        int c = idx / DPAD, k = idx % DPAD;
        mw1t[idx] = f2bf(k < 417 ? mw1[k * HID + c] : 0.0f);
        return;
    }
    idx -= 128 * DPAD;
    if (idx < 128 * 128) {
        int c = idx / 128, k = idx % 128;
        mw2t[idx] = f2bf(mw2[k * HID + c]);
        return;
    }
    idx -= 128 * 128;
    if (idx < 128 * 256) {
        int c = idx / 256, k = idx % 256;
        uw1t[idx] = f2bf(uw1[k * HID + c]);
        return;
    }
    idx -= 128 * 256;
    if (idx < 128 * 128) {
        int c = idx / 128, k = idx % 128;
        uw2t[idx] = f2bf(uw2[k * HID + c]);
    }
}

// ---------------------------------------------------------------- edge messages
// 1 block = 64 edges; 8 waves each own 16 cols x 4 M-tiles, reusing B fragments.
__global__ void edge_kernel(const float* __restrict__ h, const float* __restrict__ pos,
                            const int* __restrict__ ei, const int* __restrict__ etype,
                            const float* __restrict__ emb,
                            const float* __restrict__ mb1, const float* __restrict__ mb2,
                            const __bf16* __restrict__ mw1t, const __bf16* __restrict__ mw2t,
                            float* __restrict__ sums, float* __restrict__ counts,
                            int E) {
    __shared__ __bf16 Atile[ME][APAD];
    __shared__ __bf16 Htile[ME][HPAD];
    __shared__ float  distS[ME];
    __shared__ int    srcS[ME], dstS[ME], etS[ME];

    const int tid = threadIdx.x;
    const int eBase = blockIdx.x * ME;

    if (tid < ME) {
        int e = eBase + tid;
        int ec = e < E ? e : (E - 1);
        int s = ei[ec], d = ei[E + ec];
        srcS[tid] = s; dstS[tid] = d; etS[tid] = etype[ec];
        float dx = pos[3 * s + 0] - pos[3 * d + 0];
        float dy = pos[3 * s + 1] - pos[3 * d + 1];
        float dz = pos[3 * s + 2] - pos[3 * d + 2];
        distS[tid] = sqrtf(dx * dx + dy * dy + dz * dz);
    }
    __syncthreads();

    // Build A tile: [h_src | h_dst | emb | rbf | dist | 0-pad], float4 loads, packed LDS stores
    const float step = CUTOFF / (NUM_RBF - 1);
    const float gamma = 1.0f / (step * step);
    for (int g = tid; g < ME * (DPAD / 4); g += 256) {
        int r = g / (DPAD / 4);
        int k4 = (g % (DPAD / 4)) * 4;
        float v0, v1, v2, v3;
        if (k4 < 384) {
            const float* srcp;
            if (k4 < 128)      srcp = h   + srcS[r] * HID + k4;
            else if (k4 < 256) srcp = h   + dstS[r] * HID + (k4 - 128);
            else               srcp = emb + etS[r]  * HID + (k4 - 256);
            float4 f = *reinterpret_cast<const float4*>(srcp);
            v0 = f.x; v1 = f.y; v2 = f.z; v3 = f.w;
        } else if (k4 < 416) {
            float d0 = distS[r];
            float c0 = (float)(k4 - 384) * step;
            float a0 = d0 - c0, a1 = d0 - (c0 + step), a2 = d0 - (c0 + 2 * step), a3 = d0 - (c0 + 3 * step);
            v0 = __expf(-gamma * a0 * a0); v1 = __expf(-gamma * a1 * a1);
            v2 = __expf(-gamma * a2 * a2); v3 = __expf(-gamma * a3 * a3);
        } else if (k4 == 416) {
            v0 = distS[r]; v1 = 0.f; v2 = 0.f; v3 = 0.f;
        } else {
            v0 = v1 = v2 = v3 = 0.f;
        }
        uint2 pk; pk.x = pack2(v0, v1); pk.y = pack2(v2, v3);
        *reinterpret_cast<uint2*>(&Atile[r][k4]) = pk;
    }
    __syncthreads();

    const int lane = tid & 31;
    const int wave = tid >> 5;
    const int hrow = lane & 15;
    const int khalf = lane >> 4;
    const int colAbs = wave * 16 + (lane & 15);

    // -------- GEMM1: [64 x 448] x [448 x 128]; one B fragment feeds 4 WMMAs
    v8f acc[4] = {};
    for (int kb = 0; kb < DPAD; kb += 32) {
        v16bf b;
        const __bf16* bp = mw1t + colAbs * DPAD + kb + 16 * khalf;
#pragma unroll
        for (int j = 0; j < 16; ++j) b[j] = bp[j];
#pragma unroll
        for (int m = 0; m < 4; ++m) {
            v16bf a;
            const __bf16* ap = &Atile[16 * m + hrow][kb];
#pragma unroll
            for (int j = 0; j < 8; ++j) {
                a[j]     = ap[16 * khalf + j];
                a[8 + j] = ap[16 * khalf + 8 + j];
            }
            acc[m] = __builtin_amdgcn_wmma_f32_16x16x32_bf16(false, a, false, b, (short)0, acc[m], false, false);
        }
    }
    {
        float b1 = mb1[colAbs];
#pragma unroll
        for (int m = 0; m < 4; ++m)
#pragma unroll
            for (int v = 0; v < 8; ++v)
                Htile[16 * m + v + 8 * khalf][colAbs] = f2bf(silu(acc[m][v] + b1));
    }
    __syncthreads();

    // -------- GEMM2: [64 x 128] x [128 x 128]
    v8f acc2[4] = {};
    for (int kb = 0; kb < 128; kb += 32) {
        v16bf b;
        const __bf16* bp = mw2t + colAbs * 128 + kb + 16 * khalf;
#pragma unroll
        for (int j = 0; j < 16; ++j) b[j] = bp[j];
#pragma unroll
        for (int m = 0; m < 4; ++m) {
            v16bf a;
            const __bf16* ap = &Htile[16 * m + hrow][kb];
#pragma unroll
            for (int j = 0; j < 8; ++j) {
                a[j]     = ap[16 * khalf + j];
                a[8 + j] = ap[16 * khalf + 8 + j];
            }
            acc2[m] = __builtin_amdgcn_wmma_f32_16x16x32_bf16(false, a, false, b, (short)0, acc2[m], false, false);
        }
    }

    // -------- scatter-add messages into sums (d_out) + counts
    {
        float b2 = mb2[colAbs];
#pragma unroll
        for (int m = 0; m < 4; ++m)
#pragma unroll
            for (int v = 0; v < 8; ++v) {
                int r = 16 * m + v + 8 * khalf;
                if (eBase + r < E) {
                    float msg = silu(acc2[m][v] + b2);
                    atomicAdd(&sums[dstS[r] * HID + colAbs], msg);
                }
            }
    }
    if (tid < ME && (eBase + tid) < E) atomicAdd(&counts[dstS[tid]], 1.0f);
}

// ---------------------------------------------------------------- node update + layernorm
// 1 block = 64 nodes. sums == out (d_out): each block reads then overwrites only its own rows.
__global__ void node_kernel(const float* __restrict__ h, const int* __restrict__ node_type,
                            const float* __restrict__ ub1, const float* __restrict__ ub2,
                            const float* __restrict__ ln_g, const float* __restrict__ ln_b,
                            const __bf16* __restrict__ uw1t, const __bf16* __restrict__ uw2t,
                            const float* __restrict__ counts,
                            float* __restrict__ out,   // also holds scatter sums on entry
                            int N) {
    __shared__ __bf16 Atile[MN][NPAD];
    __shared__ __bf16 Util[MN][HPAD];
    __shared__ float  Upd[MN][UPAD];
    __shared__ float  invS[MN];

    const int tid = threadIdx.x;
    const int nBase = blockIdx.x * MN;

    if (tid < MN) {
        int node = nBase + tid; if (node >= N) node = N - 1;
        invS[tid] = 1.0f / fmaxf(counts[node], 1.0f);
    }
    __syncthreads();

    for (int g = tid; g < MN * 64; g += 256) {   // 64 float4-groups per row of 256
        int r = g / 64;
        int k4 = (g % 64) * 4;
        int node = nBase + r; if (node >= N) node = N - 1;
        float4 f;
        if (k4 < 128) {
            f = *reinterpret_cast<const float4*>(h + node * HID + k4);
        } else {
            f = *reinterpret_cast<const float4*>(out + node * HID + (k4 - 128));
            float iv = invS[r];
            f.x *= iv; f.y *= iv; f.z *= iv; f.w *= iv;
        }
        uint2 pk; pk.x = pack2(f.x, f.y); pk.y = pack2(f.z, f.w);
        *reinterpret_cast<uint2*>(&Atile[r][k4]) = pk;
    }
    __syncthreads();

    const int lane = tid & 31;
    const int wave = tid >> 5;
    const int hrow = lane & 15;
    const int khalf = lane >> 4;
    const int colAbs = wave * 16 + (lane & 15);

    // -------- GEMM1: [64 x 256] x [256 x 128]
    v8f acc[4] = {};
    for (int kb = 0; kb < 256; kb += 32) {
        v16bf b;
        const __bf16* bp = uw1t + colAbs * 256 + kb + 16 * khalf;
#pragma unroll
        for (int j = 0; j < 16; ++j) b[j] = bp[j];
#pragma unroll
        for (int m = 0; m < 4; ++m) {
            v16bf a;
            const __bf16* ap = &Atile[16 * m + hrow][kb];
#pragma unroll
            for (int j = 0; j < 8; ++j) {
                a[j]     = ap[16 * khalf + j];
                a[8 + j] = ap[16 * khalf + 8 + j];
            }
            acc[m] = __builtin_amdgcn_wmma_f32_16x16x32_bf16(false, a, false, b, (short)0, acc[m], false, false);
        }
    }
    {
        float b1 = ub1[colAbs];
#pragma unroll
        for (int m = 0; m < 4; ++m)
#pragma unroll
            for (int v = 0; v < 8; ++v)
                Util[16 * m + v + 8 * khalf][colAbs] = f2bf(silu(acc[m][v] + b1));
    }
    __syncthreads();

    // -------- GEMM2: [64 x 128] x [128 x 128] (no activation)
    v8f acc2[4] = {};
    for (int kb = 0; kb < 128; kb += 32) {
        v16bf b;
        const __bf16* bp = uw2t + colAbs * 128 + kb + 16 * khalf;
#pragma unroll
        for (int j = 0; j < 16; ++j) b[j] = bp[j];
#pragma unroll
        for (int m = 0; m < 4; ++m) {
            v16bf a;
            const __bf16* ap = &Util[16 * m + hrow][kb];
#pragma unroll
            for (int j = 0; j < 8; ++j) {
                a[j]     = ap[16 * khalf + j];
                a[8 + j] = ap[16 * khalf + 8 + j];
            }
            acc2[m] = __builtin_amdgcn_wmma_f32_16x16x32_bf16(false, a, false, b, (short)0, acc2[m], false, false);
        }
    }
    {
        float b2 = ub2[colAbs];
#pragma unroll
        for (int m = 0; m < 4; ++m)
#pragma unroll
            for (int v = 0; v < 8; ++v)
                Upd[16 * m + v + 8 * khalf][colAbs] = acc2[m][v] + b2;
    }
    __syncthreads();

    // -------- residual + LayerNorm, 8 nodes per wave, wave32 shuffle reduction
    for (int rr = 0; rr < 8; ++rr) {
        int r = wave * 8 + rr;
        int node = nBase + r;
        int nodeC = node < N ? node : (N - 1);
        float y[4], sum = 0.0f, sq = 0.0f;
#pragma unroll
        for (int c4 = 0; c4 < 4; ++c4) {
            int c = lane + 32 * c4;
            float v = h[nodeC * HID + c] + Upd[r][c];
            y[c4] = v; sum += v; sq += v * v;
        }
#pragma unroll
        for (int off = 16; off >= 1; off >>= 1) {
            sum += __shfl_xor(sum, off, 32);
            sq  += __shfl_xor(sq,  off, 32);
        }
        float mean = sum * (1.0f / 128.0f);
        float var  = sq * (1.0f / 128.0f) - mean * mean;
        float inv  = rsqrtf(var + 1e-5f);
        if (node < N) {
            bool ligand = (node_type[node] == 0);
#pragma unroll
            for (int c4 = 0; c4 < 4; ++c4) {
                int c = lane + 32 * c4;
                float res = ligand ? ((y[c4] - mean) * inv * ln_g[c] + ln_b[c])
                                   : h[node * HID + c];
                out[node * HID + c] = res;
            }
        }
    }
}

// ---------------------------------------------------------------- launch
extern "C" void kernel_launch(void* const* d_in, const int* in_sizes, int n_in,
                              void* d_out, int out_size, void* d_ws, size_t ws_size,
                              hipStream_t stream) {
    const float* h         = (const float*)d_in[0];
    const float* pos       = (const float*)d_in[1];
    const int*   ei        = (const int*)  d_in[2];
    const int*   etype     = (const int*)  d_in[3];
    const int*   node_type = (const int*)  d_in[4];
    const float* emb       = (const float*)d_in[5];
    const float* mw1       = (const float*)d_in[6];
    const float* mb1       = (const float*)d_in[7];
    const float* mw2       = (const float*)d_in[8];
    const float* mb2       = (const float*)d_in[9];
    const float* uw1       = (const float*)d_in[10];
    const float* ub1       = (const float*)d_in[11];
    const float* uw2       = (const float*)d_in[12];
    const float* ub2       = (const float*)d_in[13];
    const float* ln_g      = (const float*)d_in[14];
    const float* ln_b      = (const float*)d_in[15];

    const int N = in_sizes[0] / HID;
    const int E = in_sizes[3];

    // Workspace layout (bytes)
    char* ws = (char*)d_ws;
    __bf16* mw1t  = (__bf16*)(ws);                                   // 128*448
    __bf16* mw2t  = (__bf16*)(ws + 114688);                          // 128*128
    __bf16* uw1t  = (__bf16*)(ws + 147456);                          // 128*256
    __bf16* uw2t  = (__bf16*)(ws + 212992);                          // 128*128
    float*  counts= (float*) (ws + 245760);                          // N floats
    float*  sums  = (float*)d_out;                                   // N*128, reused as accumulator

    {
        int nSums = N * HID, nCounts = N;
        int nMax = nSums > nCounts ? nSums : nCounts;
        zero_kernel<<<(nMax + 255) / 256, 256, 0, stream>>>(sums, counts, nSums, nCounts);
    }
    {
        int total = 128 * DPAD + 128 * 128 + 128 * 256 + 128 * 128;
        prep_kernel<<<(total + 255) / 256, 256, 0, stream>>>(mw1, mw2, uw1, uw2,
                                                             mw1t, mw2t, uw1t, uw2t);
    }
    edge_kernel<<<(E + ME - 1) / ME, 256, 0, stream>>>(h, pos, ei, etype, emb, mb1, mb2,
                                                       mw1t, mw2t, sums, counts, E);
    node_kernel<<<(N + MN - 1) / MN, 256, 0, stream>>>(h, node_type, ub1, ub2, ln_g, ln_b,
                                                       uw1t, uw2t, counts, sums, N);
}
